// ImprovedActorCriticNetwork_10385230922203
// MI455X (gfx1250) — matrix operand order repaired and checked by reading, not compiled, not run on hardware
//
#include <hip/hip_runtime.h>
#include <math.h>
#include <stdint.h>

typedef __attribute__((ext_vector_type(16))) _Float16 v16h;
typedef __attribute__((ext_vector_type(8)))  _Float16 v8h;
typedef __attribute__((ext_vector_type(8)))  float    v8f;

__device__ __forceinline__ float wsum(float v) {
#pragma unroll
  for (int o = 16; o > 0; o >>= 1) v += __shfl_xor(v, o);
  return v;
}
__device__ __forceinline__ float wmaxr(float v) {
#pragma unroll
  for (int o = 16; o > 0; o >>= 1) v = fmaxf(v, __shfl_xor(v, o));
  return v;
}
__device__ __forceinline__ void atomicMaxFloat(float* addr, float val) {
  if (val >= 0.0f) atomicMax((int*)addr, __float_as_int(val));
  else             atomicMin((unsigned int*)addr, __float_as_uint(val));
}

// ---------------------------------------------------------------------------
// prep: time_enc = LN(relu([tt,0,0]@te_W+te_b)); combined (N x 8);
//       gte = LN(relu(combined[:,-3:]@gt_W+gt_b))  (N x 32)
// ---------------------------------------------------------------------------
__global__ void k_prep(const float* __restrict__ mc, const float* __restrict__ amask,
                       const float* __restrict__ spd, const float* __restrict__ dist,
                       const float* __restrict__ ttg,
                       const float* __restrict__ teW, const float* __restrict__ teb,
                       const float* __restrict__ teg, const float* __restrict__ tebe,
                       const float* __restrict__ gtW, const float* __restrict__ gtb,
                       const float* __restrict__ gtg, const float* __restrict__ gtbe,
                       float* __restrict__ comb, float* __restrict__ tenc,
                       float* __restrict__ gte, int n) {
  int i = blockIdx.x * blockDim.x + threadIdx.x;
  if (i >= n) return;
  float tt = ttg[i];
  float t0 = fmaxf(tt * teW[0] + teb[0], 0.f);
  float t1 = fmaxf(tt * teW[1] + teb[1], 0.f);
  float mu = 0.5f * (t0 + t1);
  float d0 = t0 - mu, d1 = t1 - mu;
  float rs = rsqrtf(0.5f * (d0 * d0 + d1 * d1) + 1e-5f);
  float e0 = d0 * rs * teg[0] + tebe[0];
  float e1 = d1 * rs * teg[1] + tebe[1];
  tenc[i * 2] = e0; tenc[i * 2 + 1] = e1;
  float* c = comb + (size_t)i * 8;
  c[0] = mc[i * 2]; c[1] = mc[i * 2 + 1]; c[2] = amask[i]; c[3] = spd[i];
  c[4] = dist[i];   c[5] = tt;            c[6] = e0;       c[7] = e1;
  float g[32]; float s = 0.f;
#pragma unroll
  for (int j = 0; j < 32; j++) {
    float u = tt * gtW[j] + e0 * gtW[32 + j] + e1 * gtW[64 + j] + gtb[j];
    u = fmaxf(u, 0.f); g[j] = u; s += u;
  }
  float m = s * (1.f / 32.f), v2 = 0.f;
#pragma unroll
  for (int j = 0; j < 32; j++) { float d = g[j] - m; v2 += d * d; }
  float rs2 = rsqrtf(v2 * (1.f / 32.f) + 1e-5f);
#pragma unroll
  for (int j = 0; j < 32; j++) gte[(size_t)i * 32 + j] = (g[j] - m) * rs2 * gtg[j] + gtbe[j];
}

// ---------------------------------------------------------------------------
// pack the 28 (64x64) layer weights into f16 WMMA B-fragment order:
// chunk = ((kb*4 + t)*32 + lane), 16 halfs per chunk;
// element j of chunk: K = kb*32 + (lane>=16?16:0) + j, col = t*16 + (lane&15)
// ---------------------------------------------------------------------------
__global__ void k_wprep(const float* __restrict__ gqW, const float* __restrict__ gkW,
                        const float* __restrict__ gvW, const float* __restrict__ gsW,
                        _Float16* __restrict__ wpk) {
  int tid = blockIdx.x * blockDim.x + threadIdx.x;
  if (tid >= 28 * 4096) return;
  int m = tid >> 12;          // matrix 0..27 (group-major: grp*7+layer)
  int r = tid & 4095;
  int j = r & 15;
  int chunk = r >> 4;         // 0..255
  int lane = chunk & 31;
  int kbt = chunk >> 5;       // 0..7
  int kb = kbt >> 2, t = kbt & 3;
  int K = kb * 32 + (lane >> 4) * 16 + j;
  int col = t * 16 + (lane & 15);
  int grp = m / 7, ly = m % 7;
  const float* src = (grp == 0 ? gqW : grp == 1 ? gkW : grp == 2 ? gvW : gsW)
                     + (size_t)ly * 4096;
  wpk[tid] = (_Float16)src[K * 64 + col];
}

// ---------------------------------------------------------------------------
// WMMA GEMM: Y(N x 64) = Xh(N x 64, f16) @ W(64 x 64) + b.
// 256 threads = 8 waves, each wave -> 16x64 output tile (block: 128 rows).
// Packed f16 weights (8KB) staged to LDS via global_load_async_to_lds_b128.
// A frag (16x32 f16): lane<16 row=l: K0..15 ; lane>=16: K16..31
// B frag (32x16 f16): lane<16 col: K0..15 ; lane>=16 col: K16..31
// C/D: VGPR r -> (M=r, N=lane) lanes0-15 ; (M=r+8, N=lane-16) lanes16-31
// ---------------------------------------------------------------------------
__global__ __launch_bounds__(256)
void k_gemm64_wmma(const _Float16* __restrict__ Xh, const _Float16* __restrict__ Wp,
                   const float* __restrict__ B, float* __restrict__ Y) {
  __shared__ __align__(32) _Float16 wlds[4096];
  int tid = threadIdx.x;

  // async-stage 8KB of packed weights into LDS (32B per thread, 2x b128)
  {
    unsigned      l = (unsigned)(uintptr_t)(&wlds[0]) + (unsigned)tid * 32u;
    unsigned long g = (unsigned long)(uintptr_t)Wp + (unsigned long)tid * 32ul;
    asm volatile("global_load_async_to_lds_b128 %0, %1, off"
                 :: "v"(l), "v"(g) : "memory");
    unsigned      l2 = l + 16u;
    unsigned long g2 = g + 16ul;
    asm volatile("global_load_async_to_lds_b128 %0, %1, off"
                 :: "v"(l2), "v"(g2) : "memory");
    asm volatile("s_wait_asynccnt 0" ::: "memory");
  }
  __syncthreads();

  int lane = tid & 31;
  int wv   = tid >> 5;
  int hs   = lane >> 4;
  int idx  = lane & 15;
  size_t row0 = (size_t)blockIdx.x * 128 + (size_t)wv * 16;
  const _Float16* xrow = Xh + (row0 + idx) * 64;

  v8f acc[4] = {v8f{}, v8f{}, v8f{}, v8f{}};

#pragma unroll
  for (int kb = 0; kb < 2; kb++) {
    int k0 = kb * 32 + hs * 16;
    int k1 = kb * 32 + hs * 16 + 8;
    v8h a0 = *(const v8h*)(xrow + k0);
    v8h a1 = *(const v8h*)(xrow + k1);
    v16h a;
#pragma unroll
    for (int j = 0; j < 8; j++) { a[j] = a0[j]; a[8 + j] = a1[j]; }
#pragma unroll
    for (int t = 0; t < 4; t++) {
      const _Float16* wc = &wlds[((kb * 4 + t) * 32 + lane) * 16];
      v8h b0 = *(const v8h*)wc;
      v8h b1 = *(const v8h*)(wc + 8);
      v16h bf;
#pragma unroll
      for (int j = 0; j < 8; j++) { bf[j] = b0[j]; bf[8 + j] = b1[j]; }
      acc[t] = __builtin_amdgcn_wmma_f32_16x16x32_f16(
          false, a, false, bf, (short)0, acc[t], false, false);
    }
  }
#pragma unroll
  for (int t = 0; t < 4; t++) {
    int col = t * 16 + idx;
    float bias = B[col];
#pragma unroll
    for (int r = 0; r < 8; r++) {
      size_t row = row0 + r + hs * 8;
      Y[row * 64 + col] = acc[t][r] + bias;
    }
  }
}

// ---------------------------------------------------------------------------
// layer-0 GEMM: Y(N x 64) = X(N x 8) @ W(8 x 64) + b
// ---------------------------------------------------------------------------
__global__ void k_gemm8(const float* __restrict__ X, const float* __restrict__ Wm,
                        const float* __restrict__ B, float* __restrict__ Y, int n) {
  int idx = blockIdx.x * blockDim.x + threadIdx.x;
  if (idx >= n * 64) return;
  int i = idx >> 6, c = idx & 63;
  const float* x = X + (size_t)i * 8;
  float acc = B[c];
#pragma unroll
  for (int k = 0; k < 8; k++) acc += x[k] * Wm[k * 64 + c];
  Y[idx] = acc;
}

// ---------------------------------------------------------------------------
// edge attention (segment softmax) kernels
// ---------------------------------------------------------------------------
__global__ void k_attn_init(float* __restrict__ agg, float* __restrict__ nmax,
                            float* __restrict__ nden, int n) {
  int idx = blockIdx.x * blockDim.x + threadIdx.x;
  if (idx >= n * 64) return;
  agg[idx] = 0.f;
  if (idx < n * 2) { nmax[idx] = -INFINITY; nden[idx] = 0.f; }
}

__global__ void k_edge_logits(const int* __restrict__ ei, const float* __restrict__ qb,
                              const float* __restrict__ kb, float* __restrict__ elog,
                              float* __restrict__ nmax, int E_) {
  int idx = blockIdx.x * blockDim.x + threadIdx.x;
  if (idx >= E_ * 2) return;
  int e = idx >> 1, h = idx & 1;
  int s = ei[e], d = ei[E_ + e];
  const float* q = qb + (size_t)d * 64 + h * 32;
  const float* k = kb + (size_t)s * 64 + h * 32;
  float acc = 0.f;
#pragma unroll
  for (int g = 0; g < 32; g++) acc += q[g] * k[g];
  acc *= 0.17677669529663687f; // 1/sqrt(32)
  elog[idx] = acc;
  atomicMaxFloat(&nmax[d * 2 + h], acc);
}

__global__ void k_edge_exp(const int* __restrict__ ei, float* __restrict__ elog,
                           const float* __restrict__ nmax, float* __restrict__ nden, int E_) {
  int idx = blockIdx.x * blockDim.x + threadIdx.x;
  if (idx >= E_ * 2) return;
  int e = idx >> 1, h = idx & 1;
  int d = ei[E_ + e];
  float ex = expf(elog[idx] - nmax[d * 2 + h]);
  elog[idx] = ex;
  atomicAdd(&nden[d * 2 + h], ex);
}

__global__ void k_edge_agg(const int* __restrict__ ei, const float* __restrict__ elog,
                           const float* __restrict__ nden, const float* __restrict__ vb,
                           float* __restrict__ agg, int E_) {
  int idx = blockIdx.x * blockDim.x + threadIdx.x;
  if (idx >= E_ * 64) return;
  int e = idx >> 6, c = idx & 63, h = c >> 5;
  int s = ei[e], d = ei[E_ + e];
  float alpha = elog[e * 2 + h] / nden[d * 2 + h];
  atomicAdd(&agg[(size_t)d * 64 + c], vb[(size_t)s * 64 + c] * alpha);
}

// combine: cur = relu(agg + s) + res, plus f16 mirror for the next WMMA GEMM
__global__ void k_combine(const float* __restrict__ agg, const float* __restrict__ sb,
                          const float* __restrict__ res, float* __restrict__ cur,
                          _Float16* __restrict__ curh, int n) {
  int idx = blockIdx.x * blockDim.x + threadIdx.x;
  if (idx >= n * 64) return;
  float v = fmaxf(agg[idx] + sb[idx], 0.f) + res[idx];
  cur[idx] = v;
  curh[idx] = (_Float16)v;
}

// ---------------------------------------------------------------------------
// gfu = LN(relu([cur, gte] @ gf_W + gf_b)); gnn_out = gfu @ go_W + go_b
// ---------------------------------------------------------------------------
__global__ __launch_bounds__(32)
void k_gf(const float* __restrict__ cur, const float* __restrict__ gte,
          const float* __restrict__ gfW, const float* __restrict__ gfb,
          const float* __restrict__ gfg, const float* __restrict__ gfbe,
          const float* __restrict__ goW, const float* __restrict__ gob,
          float* __restrict__ gnn) {
  int i = blockIdx.x, j = threadIdx.x;
  const float* xr = cur + (size_t)i * 64;
  const float* gr = gte + (size_t)i * 32;
  float u = gfb[j];
#pragma unroll 8
  for (int t = 0; t < 64; t++) u += xr[t] * gfW[t * 32 + j];
#pragma unroll 8
  for (int t = 0; t < 32; t++) u += gr[t] * gfW[(64 + t) * 32 + j];
  u = fmaxf(u, 0.f);
  float m = wsum(u) * (1.f / 32.f);
  float d = u - m;
  float var = wsum(d * d) * (1.f / 32.f);
  float rs = rsqrtf(var + 1e-5f);
  float gf = d * rs * gfg[j] + gfbe[j];
  float p0 = wsum(gf * goW[j * 2]);
  float p1 = wsum(gf * goW[j * 2 + 1]);
  if (j == 0) { gnn[i * 2] = p0 + gob[0]; gnn[i * 2 + 1] = p1 + gob[1]; }
}

// ---------------------------------------------------------------------------
// tiny MHA on time_enc (EMB=2, H=2, hd=1)
// ---------------------------------------------------------------------------
__global__ void k_mha_qkv(const float* __restrict__ tenc,
                          const float* __restrict__ qW, const float* __restrict__ qb,
                          const float* __restrict__ kW, const float* __restrict__ kb,
                          const float* __restrict__ vW, const float* __restrict__ vb,
                          float* __restrict__ mq, float* __restrict__ mk,
                          float* __restrict__ mv, int n) {
  int i = blockIdx.x * blockDim.x + threadIdx.x;
  if (i >= n) return;
  float x0 = tenc[i * 2], x1 = tenc[i * 2 + 1];
#pragma unroll
  for (int j = 0; j < 2; j++) {
    mq[i * 2 + j] = x0 * qW[j] + x1 * qW[2 + j] + qb[j];
    mk[i * 2 + j] = x0 * kW[j] + x1 * kW[2 + j] + kb[j];
    mv[i * 2 + j] = x0 * vW[j] + x1 * vW[2 + j] + vb[j];
  }
}

__global__ __launch_bounds__(32)
void k_mha(const float* __restrict__ mq, const float* __restrict__ mk,
           const float* __restrict__ mv, const float* __restrict__ moW,
           const float* __restrict__ mob, float* __restrict__ tctx, int n) {
  int i = blockIdx.x, lane = threadIdx.x;
  float q0 = mq[i * 2], q1 = mq[i * 2 + 1];
  float m0 = -INFINITY, m1 = -INFINITY;
  for (int k = lane; k < n; k += 32) {
    m0 = fmaxf(m0, q0 * mk[k * 2]);
    m1 = fmaxf(m1, q1 * mk[k * 2 + 1]);
  }
  m0 = wmaxr(m0); m1 = wmaxr(m1);
  float s0 = 0.f, s1 = 0.f, o0 = 0.f, o1 = 0.f;
  for (int k = lane; k < n; k += 32) {
    float e0 = expf(q0 * mk[k * 2] - m0);     s0 += e0; o0 += e0 * mv[k * 2];
    float e1 = expf(q1 * mk[k * 2 + 1] - m1); s1 += e1; o1 += e1 * mv[k * 2 + 1];
  }
  s0 = wsum(s0); s1 = wsum(s1); o0 = wsum(o0); o1 = wsum(o1);
  if (lane == 0) {
    float a0 = o0 / s0, a1 = o1 / s1;
    tctx[i * 2]     = a0 * moW[0] + a1 * moW[2] + mob[0];
    tctx[i * 2 + 1] = a0 * moW[1] + a1 * moW[3] + mob[1];
  }
}

// ---------------------------------------------------------------------------
// fused = LN(relu([gnn, tenc, tctx] @ ff_W + ff_b))  (N x 2)
// ---------------------------------------------------------------------------
__global__ void k_fused(const float* __restrict__ gnn, const float* __restrict__ tenc,
                        const float* __restrict__ tctx, const float* __restrict__ ffW,
                        const float* __restrict__ ffb, const float* __restrict__ ffg,
                        const float* __restrict__ ffbe, float* __restrict__ fus, int n) {
  int i = blockIdx.x * blockDim.x + threadIdx.x;
  if (i >= n) return;
  float in6[6] = {gnn[i * 2], gnn[i * 2 + 1], tenc[i * 2], tenc[i * 2 + 1],
                  tctx[i * 2], tctx[i * 2 + 1]};
  float u0 = ffb[0], u1 = ffb[1];
#pragma unroll
  for (int k = 0; k < 6; k++) { u0 += in6[k] * ffW[k * 2]; u1 += in6[k] * ffW[k * 2 + 1]; }
  u0 = fmaxf(u0, 0.f); u1 = fmaxf(u1, 0.f);
  float mu = 0.5f * (u0 + u1);
  float d0 = u0 - mu, d1 = u1 - mu;
  float rs = rsqrtf(0.5f * (d0 * d0 + d1 * d1) + 1e-5f);
  fus[i * 2]     = d0 * rs * ffg[0] + ffbe[0];
  fus[i * 2 + 1] = d1 * rs * ffg[1] + ffbe[1];
}

// ---------------------------------------------------------------------------
// actor + critic heads; one 128-thread block per node
// ---------------------------------------------------------------------------
__global__ __launch_bounds__(128)
void k_heads(const float* __restrict__ fus,
             const float* __restrict__ asW, const float* __restrict__ asb,
             const float* __restrict__ asg, const float* __restrict__ asbe,
             const float* __restrict__ aoW, const float* __restrict__ aob,
             const float* __restrict__ c1W, const float* __restrict__ c1b,
             const float* __restrict__ c1g, const float* __restrict__ c1be,
             const float* __restrict__ c2W, const float* __restrict__ c2b,
             float* __restrict__ out, int n) {
  int i = blockIdx.x, t = threadIdx.x;
  float f0 = fus[i * 2], f1 = fus[i * 2 + 1];
  float a = fmaxf(f0 * asW[t] + f1 * asW[128 + t] + asb[t], 0.f);
  float c = fmaxf(f0 * c1W[t] + f1 * c1W[128 + t] + c1b[t], 0.f);
  __shared__ float sa[128], sc[128], st[4];
  sa[t] = a; sc[t] = c;
  __syncthreads();
  if (t == 0) {
    float s = 0.f; for (int j = 0; j < 128; j++) s += sa[j];
    float m = s * (1.f / 128.f), v = 0.f;
    for (int j = 0; j < 128; j++) { float d = sa[j] - m; v += d * d; }
    st[0] = m; st[1] = rsqrtf(v * (1.f / 128.f) + 1e-5f);
  }
  if (t == 1) {
    float s = 0.f; for (int j = 0; j < 128; j++) s += sc[j];
    float m = s * (1.f / 128.f), v = 0.f;
    for (int j = 0; j < 128; j++) { float d = sc[j] - m; v += d * d; }
    st[2] = m; st[3] = rsqrtf(v * (1.f / 128.f) + 1e-5f);
  }
  __syncthreads();
  float af = (a - st[0]) * st[1] * asg[t] + asbe[t];
  float cf = (c - st[2]) * st[3] * c1g[t] + c1be[t];
  sa[t] = af; sc[t] = cf;
  __syncthreads();
  if (t < 64) {
    float lg = aob[t];
#pragma unroll 8
    for (int j = 0; j < 128; j++) lg += sa[j] * aoW[j * 64 + t];
    out[(size_t)i * 64 + t] = lg;
  }
  if (t == 127) {
    float val = c2b[0];
    for (int j = 0; j < 128; j++) val += sc[j] * c2W[j];
    out[(size_t)n * 64 + i] = val;
  }
}

// ---------------------------------------------------------------------------
extern "C" void kernel_launch(void* const* d_in, const int* in_sizes, int n_in,
                              void* d_out, int out_size, void* d_ws, size_t ws_size,
                              hipStream_t stream) {
  const int n  = in_sizes[7];      // N
  const int E_ = in_sizes[1] / 2;  // E

  auto f = [&](int i) { return (const float*)d_in[i]; };
  const float* mission = f(0);
  const int*   ei      = (const int*)d_in[1];
  const float* amask = f(4); const float* spd = f(5);
  const float* dist  = f(6); const float* ttg = f(7);
  const float* teW = f(8),  *teb = f(9),  *teg = f(10), *tebe = f(11);
  const float* mqW = f(12), *mqb = f(13), *mkW = f(14), *mkb = f(15);
  const float* mvW = f(16), *mvb = f(17), *moW = f(18), *mob = f(19);
  const float* ffW = f(20), *ffb = f(21), *ffg = f(22), *ffbe = f(23);
  const float* asW = f(24), *asb = f(25), *asg = f(26), *asbe = f(27);
  const float* aoW = f(28), *aob = f(29);
  const float* c1W = f(30), *c1b = f(31), *c1g = f(32), *c1be = f(33);
  const float* c2W = f(34), *c2b = f(35);
  const float* gtW = f(36), *gtb = f(37), *gtg = f(38), *gtbe = f(39);
  const float* resW = f(40), *resb = f(41);
  const float* gq0W = f(42), *gq0b = f(43), *gk0W = f(44), *gk0b = f(45);
  const float* gv0W = f(46), *gv0b = f(47), *gs0W = f(48), *gs0b = f(49);
  const float* gqW = f(50), *gqb = f(51), *gkW = f(52), *gkb = f(53);
  const float* gvW = f(54), *gvb = f(55), *gsW = f(56), *gsb = f(57);
  const float* gfW = f(58), *gfb = f(59), *gfg = f(60), *gfbe = f(61);
  const float* goW = f(62), *gob = f(63);

  float* W = (float*)d_ws;
  size_t o = 0;
  float* comb = W + o; o += (size_t)n * 8;
  float* tenc = W + o; o += (size_t)n * 2;
  float* gte  = W + o; o += (size_t)n * 32;
  float* cur  = W + o; o += (size_t)n * 64;
  float* res  = W + o; o += (size_t)n * 64;
  float* qb   = W + o; o += (size_t)n * 64;
  float* kb   = W + o; o += (size_t)n * 64;
  float* vb   = W + o; o += (size_t)n * 64;
  float* sb   = W + o; o += (size_t)n * 64;
  float* agg  = W + o; o += (size_t)n * 64;
  float* elog = W + o; o += (size_t)E_ * 2;
  float* nmax = W + o; o += (size_t)n * 2;
  float* nden = W + o; o += (size_t)n * 2;
  float* mq   = W + o; o += (size_t)n * 2;
  float* mk   = W + o; o += (size_t)n * 2;
  float* mv   = W + o; o += (size_t)n * 2;
  float* tctx = W + o; o += (size_t)n * 2;
  float* gnn  = W + o; o += (size_t)n * 2;
  float* fus  = W + o; o += (size_t)n * 2;
  _Float16* curh = (_Float16*)(W + o); o += (size_t)n * 32;   // N*64 halfs
  _Float16* wpk  = (_Float16*)(W + o); o += (size_t)28 * 2048; // 28*4096 halfs

  const int TB = 256;
  dim3 b(TB);
  int gN    = (n + TB - 1) / TB;
  int gN64  = (n * 64 + TB - 1) / TB;
  int gE2   = (E_ * 2 + TB - 1) / TB;
  int gE64  = (int)(((size_t)E_ * 64 + TB - 1) / TB);

  // one-time per launch: pack layer weights to f16 fragment layout
  k_wprep<<<(28 * 4096 + TB - 1) / TB, b, 0, stream>>>(gqW, gkW, gvW, gsW, wpk);

  k_prep<<<gN, b, 0, stream>>>(mission, amask, spd, dist, ttg, teW, teb, teg, tebe,
                               gtW, gtb, gtg, gtbe, comb, tenc, gte, n);

  auto run_attn = [&]() {
    k_attn_init<<<gN64, b, 0, stream>>>(agg, nmax, nden, n);
    k_edge_logits<<<gE2, b, 0, stream>>>(ei, qb, kb, elog, nmax, E_);
    k_edge_exp<<<gE2, b, 0, stream>>>(ei, elog, nmax, nden, E_);
    k_edge_agg<<<gE64, b, 0, stream>>>(ei, elog, nden, vb, agg, E_);
  };

  // ---- layer 0 (K=8 scalar GEMMs) ----
  k_gemm8<<<gN64, b, 0, stream>>>(comb, resW, resb, res, n);
  k_gemm8<<<gN64, b, 0, stream>>>(comb, gq0W, gq0b, qb, n);
  k_gemm8<<<gN64, b, 0, stream>>>(comb, gk0W, gk0b, kb, n);
  k_gemm8<<<gN64, b, 0, stream>>>(comb, gv0W, gv0b, vb, n);
  k_gemm8<<<gN64, b, 0, stream>>>(comb, gs0W, gs0b, sb, n);
  run_attn();
  k_combine<<<gN64, b, 0, stream>>>(agg, sb, res, cur, curh, n);

  // ---- layers 1..7 (WMMA GEMMs, async-LDS staged weights) ----
  int gW = n / 128;   // 8 waves per block, 16 rows per wave
  for (int l = 1; l < 8; l++) {
    size_t ly = (size_t)(l - 1);
    size_t bo = ly * 64;
    k_gemm64_wmma<<<gW, 256, 0, stream>>>(curh, wpk + (0 * 7 + ly) * 4096, gqb + bo, qb);
    k_gemm64_wmma<<<gW, 256, 0, stream>>>(curh, wpk + (1 * 7 + ly) * 4096, gkb + bo, kb);
    k_gemm64_wmma<<<gW, 256, 0, stream>>>(curh, wpk + (2 * 7 + ly) * 4096, gvb + bo, vb);
    k_gemm64_wmma<<<gW, 256, 0, stream>>>(curh, wpk + (3 * 7 + ly) * 4096, gsb + bo, sb);
    run_attn();
    k_combine<<<gN64, b, 0, stream>>>(agg, sb, cur, cur, curh, n); // res = cur
  }

  // ---- GNN output head ----
  k_gf<<<n, 32, 0, stream>>>(cur, gte, gfW, gfb, gfg, gfbe, goW, gob, gnn);

  // ---- tiny MHA over time_enc ----
  k_mha_qkv<<<gN, b, 0, stream>>>(tenc, mqW, mqb, mkW, mkb, mvW, mvb, mq, mk, mv, n);
  k_mha<<<n, 32, 0, stream>>>(mq, mk, mv, moW, mob, tctx, n);

  // ---- fusion + actor/critic heads ----
  k_fused<<<gN, b, 0, stream>>>(gnn, tenc, tctx, ffW, ffb, ffg, ffbe, fus, n);
  k_heads<<<n, 128, 0, stream>>>(fus, asW, asb, asg, asbe, aoW, aob,
                                 c1W, c1b, c1g, c1be, c2W, c2b, (float*)d_out, n);
}
